// SemanticConflictDetector_58076547776526
// MI455X (gfx1250) — matrix operand order repaired and checked by reading, not compiled, not run on hardware
//
#include <hip/hip_runtime.h>
#include <math.h>

// ---------------- problem constants ----------------
#define ROWS     32      // batch rows per workgroup (two 16-row WMMA M-tiles)
#define ED       512     // EMBED_DIM
#define NH       8       // NUM_HEADS
#define HD       64      // HEAD_DIM
#define HIDW     128     // HID
#define NC1      1024    // NH*HIDW  (all heads' GEMM1 columns)
#define KCH      32      // K chunk per v_wmma_f32_16x16x32_bf16
#define NTHREADS 512     // 16 wave32 waves: waves 0-7 -> M-tile 0, 8-15 -> M-tile 1

// fragment-linear weight workspace (bf16 halves)
#define W1F_HALFS  (64 * 16 * 32 * 16)   // [ntile=64][kk=16][lane=32][16]  = 1 MB
#define GW1F_HALFS (32 * 32 * 32 * 16)   // [ntile=32][kk=32][lane=32][16]  = 1 MB
#define W2F_HALFS  (8 * 4 * 4 * 32 * 16) // [head][t=4][kq=4][lane=32][16]  = 128 KB
#define WS_HALFS   (W1F_HALFS + GW1F_HALFS + W2F_HALFS)

typedef __bf16 bf16_t;
typedef bf16_t bf16x4  __attribute__((ext_vector_type(4)));
typedef bf16_t bf16x8  __attribute__((ext_vector_type(8)));
typedef bf16_t bf16x16 __attribute__((ext_vector_type(16)));
typedef float  f32x4   __attribute__((ext_vector_type(4)));
typedef float  f32x8   __attribute__((ext_vector_type(8)));

union BFU { bf16x16 v; struct { bf16x8 lo, hi; } p; };

// A fragment (16x32 bf16, M x K), contiguous-K layout used by this file (same order as the B packing):
//   lane = m + 16*khalf ; halves 0..7  = K (k0 + khalf*16 + 0..7)
//                         halves 8..15 = K (k0 + khalf*16 + 8 + 0..7)
static __device__ __forceinline__
bf16x16 load_a_frag(const bf16_t* __restrict__ base, int stride, int m, int khalf, int k0) {
  BFU u;
  u.p.lo = *(const bf16x8*)(base + m * stride + k0 + khalf * 16);
  u.p.hi = *(const bf16x8*)(base + m * stride + k0 + khalf * 16 + 8);
  return u.v;
}

// B fragment from fragment-linear global memory: lane's 16 halves are contiguous
static __device__ __forceinline__
bf16x16 load_frag_g(const bf16_t* __restrict__ p) {
  BFU u;
  u.p.lo = *(const bf16x8*)(p);
  u.p.hi = *(const bf16x8*)(p + 8);
  return u.v;
}

static __device__ __forceinline__ float gelu_erf(float x) {
  return 0.5f * x * (1.0f + erff(x * 0.7071067811865476f));
}

// ---------------- weight prep: f32 -> bf16 fragment-linear layout in d_ws ----------------
// B-frag layout assumption: lane = n + 16*khalf ; halves 0..15 = K (khalf*16 + 0..15)
__global__ void prep_weights(const float* __restrict__ W1, const float* __restrict__ W2,
                             const float* __restrict__ gW1, bf16_t* __restrict__ ws) {
  long id = (long)blockIdx.x * blockDim.x + threadIdx.x;
  if (id < W1F_HALFS) {
    int h = (int)id & 15, lane = (int)(id >> 4) & 31, kk = (int)(id >> 9) & 15, nt = (int)(id >> 13);
    int n = nt * 16 + (lane & 15);
    int k = kk * KCH + (lane >> 4) * 16 + h;
    int head = n >> 7, kc = n & (HIDW - 1);
    ws[id] = (bf16_t)W1[((long)head * ED + k) * HIDW + kc];
  } else if (id < W1F_HALFS + GW1F_HALFS) {
    long j = id - W1F_HALFS;
    int h = (int)j & 15, lane = (int)(j >> 4) & 31, kk = (int)(j >> 9) & 31, nt = (int)(j >> 14);
    int n = nt * 16 + (lane & 15);
    int k = kk * KCH + (lane >> 4) * 16 + h;
    ws[id] = (bf16_t)gW1[(long)k * ED + n];
  } else if (id < WS_HALFS) {
    long j = id - W1F_HALFS - GW1F_HALFS;
    int h = (int)j & 15, lane = (int)(j >> 4) & 31, kq = (int)(j >> 9) & 3,
        t = (int)(j >> 11) & 3, head = (int)(j >> 13);
    int ncol = t * 16 + (lane & 15);
    int k = kq * KCH + (lane >> 4) * 16 + h;
    ws[id] = (bf16_t)W2[((long)head * HIDW + k) * HD + ncol];
  }
}

// ---------------- fused main kernel ----------------
__global__ __launch_bounds__(NTHREADS, 1)
void fused_conflict_kernel(
    const float* __restrict__ Xt,  const float* __restrict__ Xi,
    const float* __restrict__ b1,
    const float* __restrict__ lg1, const float* __restrict__ lb1,
    const float* __restrict__ b2,
    const float* __restrict__ aw,
    const float* __restrict__ gb1,
    const float* __restrict__ glg, const float* __restrict__ glb,
    const float* __restrict__ gW2, const float* __restrict__ gb2,
    const bf16_t* __restrict__ ws,
    float* __restrict__ out_conf,  float* __restrict__ out_pa)
{
  extern __shared__ char smem[];
  bf16_t* sXt = (bf16_t*)smem;                 // [32][512] bf16  32KB
  bf16_t* sXi = sXt + ROWS * ED;               // [32][512] bf16  32KB
  bf16_t* sHn = sXi + ROWS * ED;               // [32][1024] bf16 64KB (GEMM2 A input)
  float*  sPA = (float*)(sHn + ROWS * NC1);    // [32][8]          1KB
  // sG ALIASES sXt+sXi: the concat inputs are dead once the global-proj
  // K-sweep completes (barrier below makes the overlay safe).
  float*  sG  = (float*)smem;                  // [32][512] f32   64KB
  // total footprint ~129 KB -> two workgroups can co-reside per WGP (258 KB)

  const bf16_t* W1F  = ws;
  const bf16_t* gW1F = ws + W1F_HALFS;
  const bf16_t* W2F  = ws + W1F_HALFS + GW1F_HALFS;

  const int tid   = threadIdx.x;
  const int wave  = tid >> 5;
  const int lane  = tid & 31;
  const int m     = lane & 15;
  const int khalf = lane >> 4;
  const int mtile = wave >> 3;       // which 16-row M-tile (0..1)
  const int head  = wave & 7;        // head owned by this wave
  const long row0 = (long)blockIdx.x * ROWS;

  // ---- stage both embedding tiles to LDS as bf16 (vectorized, coalesced) ----
  for (int e4 = tid; e4 < ROWS * ED / 4; e4 += NTHREADS) {
    f32x4 t4 = *(const f32x4*)(Xt + row0 * ED + e4 * 4);
    f32x4 i4 = *(const f32x4*)(Xi + row0 * ED + e4 * 4);
    bf16x4 tb, ib;
#pragma unroll
    for (int q = 0; q < 4; ++q) { tb[q] = (bf16_t)t4[q]; ib[q] = (bf16_t)i4[q]; }
    *(bf16x4*)(sXt + e4 * 4) = tb;
    *(bf16x4*)(sXi + e4 * 4) = ib;
  }
  __syncthreads();

  const bf16_t* sXtM = sXt + mtile * 16 * ED;
  const bf16_t* sXiM = sXi + mtile * 16 * ED;

  // ---- GEMM1: [16,512] x [512,128] (head slice), text AND image, one K-sweep ----
  f32x8 accT[8] = {};
  f32x8 accI[8] = {};
  for (int kk = 0; kk < ED / KCH; ++kk) {
    bf16x16 at = load_a_frag(sXtM, ED, m, khalf, kk * KCH);
    bf16x16 ai = load_a_frag(sXiM, ED, m, khalf, kk * KCH);
    bf16x16 bfr[8];
#pragma unroll
    for (int t = 0; t < 8; ++t)
      bfr[t] = load_frag_g(W1F + ((((long)(head * 8 + t) * 16 + kk) * 32 + lane) * 16));
#pragma unroll
    for (int t = 0; t < 8; ++t) {
      accT[t] = __builtin_amdgcn_wmma_f32_16x16x32_bf16(false, at, false, bfr[t], (short)0, accT[t], false, false);
      accI[t] = __builtin_amdgcn_wmma_f32_16x16x32_bf16(false, ai, false, bfr[t], (short)0, accI[t], false, false);
    }
  }

  // normalized text head-vectors live in registers between the two passes
  f32x8 tn[4];

  // ---- per-input tail, all statistics in-register (wave owns its whole head) ----
  auto head_tail = [&](f32x8 (&acc)[8], int inp) {
    // + b1, then per-row mean/var over the 128 head columns (8 tiles x 16 lanes)
#pragma unroll
    for (int t = 0; t < 8; ++t) {
      float bias = b1[head * HIDW + t * 16 + m];
#pragma unroll
      for (int r = 0; r < 8; ++r) acc[t][r] += bias;
    }
    float mu[8], rs[8];
#pragma unroll
    for (int r = 0; r < 8; ++r) {
      float s1 = 0.f, s2 = 0.f;
#pragma unroll
      for (int t = 0; t < 8; ++t) { float v = acc[t][r]; s1 += v; s2 += v * v; }
#pragma unroll
      for (int off = 8; off; off >>= 1) { s1 += __shfl_xor(s1, off); s2 += __shfl_xor(s2, off); }
      float m1 = s1 * (1.f / HIDW);
      mu[r] = m1;
      rs[r] = rsqrtf(s2 * (1.f / HIDW) - m1 * m1 + 1e-5f);
    }
    // LN scale/shift + exact-erf GELU -> sHn (bf16), wave-local region
#pragma unroll
    for (int t = 0; t < 8; ++t) {
      int cc = head * HIDW + t * 16 + m;
      float gg = lg1[cc], bb = lb1[cc];
#pragma unroll
      for (int r = 0; r < 8; ++r) {
        float v = (acc[t][r] - mu[r]) * rs[r] * gg + bb;
        sHn[(mtile * 16 + r + 8 * khalf) * NC1 + cc] = (bf16_t)gelu_erf(v);
      }
    }
    __syncthreads();   // cheap; guarantees sHn visibility for the A-fragment reads

    // GEMM2: [16,128] x [128,64] for this head (4 N-tiles, K=128)
    f32x8 o[4] = {};
#pragma unroll
    for (int kq = 0; kq < 4; ++kq) {
      bf16x16 a = load_a_frag(sHn + mtile * 16 * NC1 + head * HIDW, NC1, m, khalf, kq * KCH);
#pragma unroll
      for (int t = 0; t < 4; ++t) {
        bf16x16 b = load_frag_g(W2F + ((((long)(head * 4 + t) * 4 + kq) * 32 + lane) * 16));
        o[t] = __builtin_amdgcn_wmma_f32_16x16x32_bf16(false, a, false, b, (short)0, o[t], false, false);
      }
    }
#pragma unroll
    for (int t = 0; t < 4; ++t) {
      float bias = b2[head * HD + t * 16 + m];
#pragma unroll
      for (int r = 0; r < 8; ++r) o[t][r] += bias;
    }
    // row-wise L2 norm over 64 cols (4 tiles x 16 lanes); text stores to regs,
    // image dots against the register-resident normalized text
#pragma unroll
    for (int r = 0; r < 8; ++r) {
      float ss = 0.f;
#pragma unroll
      for (int t = 0; t < 4; ++t) ss += o[t][r] * o[t][r];
#pragma unroll
      for (int off = 8; off; off >>= 1) ss += __shfl_xor(ss, off);
      float inv = 1.f / fmaxf(sqrtf(ss), 1e-12f);
      if (inp == 0) {
#pragma unroll
        for (int t = 0; t < 4; ++t) tn[t][r] = o[t][r] * inv;
      } else {
        float d = 0.f;
#pragma unroll
        for (int t = 0; t < 4; ++t) d += o[t][r] * inv * tn[t][r];
#pragma unroll
        for (int off = 8; off; off >>= 1) d += __shfl_xor(d, off);
        if (m == 0) {
          int row = mtile * 16 + r + 8 * khalf;
          sPA[row * NH + head] = d;
          out_pa[(row0 + row) * NH + head] = d;
        }
      }
    }
  };
  head_tail(accT, 0);
  head_tail(accI, 1);

  // ---- global proj GEMM: concat[16,1024] x gW1[1024,512] per M-tile ----
  f32x8 accG[4] = {};
  for (int kk = 0; kk < (2 * ED) / KCH; ++kk) {
    const bf16_t* sX = (kk < ED / KCH) ? sXtM : sXiM;   // concat([text,image])
    bf16x16 a = load_a_frag(sX, ED, m, khalf, (kk * KCH) & (ED - 1));
    bf16x16 bfr[4];
#pragma unroll
    for (int t = 0; t < 4; ++t)
      bfr[t] = load_frag_g(gW1F + ((((long)(head * 4 + t) * 32 + kk) * 32 + lane) * 16));
#pragma unroll
    for (int t = 0; t < 4; ++t)
      accG[t] = __builtin_amdgcn_wmma_f32_16x16x32_bf16(false, a, false, bfr[t], (short)0, accG[t], false, false);
  }
  __syncthreads();   // all reads of sXt/sXi complete -> safe to overwrite with sG
#pragma unroll
  for (int t = 0; t < 4; ++t) {
    int n = head * 64 + t * 16 + m;
    float bias = gb1[n];
#pragma unroll
    for (int r = 0; r < 8; ++r)
      sG[(mtile * 16 + r + 8 * khalf) * ED + n] = accG[t][r] + bias;
  }
  __syncthreads();

  // ---- LN(512) + GELU + dot(gW2) + sigmoid + final combine; wave per row ----
  float wsm[NH];
  {
    float mx = aw[0];
    for (int h = 1; h < NH; ++h) mx = fmaxf(mx, aw[h]);
    float s = 0.f;
    for (int h = 0; h < NH; ++h) { wsm[h] = expf(aw[h] - mx); s += wsm[h]; }
    float inv = 1.f / s;
    for (int h = 0; h < NH; ++h) wsm[h] *= inv;
  }
  for (int rr = 0; rr < 2; ++rr) {
    int r = wave + rr * 16;
    const float* x = sG + r * ED;
    float s1 = 0.f, s2 = 0.f;
    for (int c = lane; c < ED; c += 32) { float v = x[c]; s1 += v; s2 += v * v; }
    for (int off = 16; off; off >>= 1) { s1 += __shfl_xor(s1, off); s2 += __shfl_xor(s2, off); }
    float mu = s1 * (1.f / ED);
    float var = s2 * (1.f / ED) - mu * mu;
    float rsg = rsqrtf(var + 1e-5f);
    float dot = 0.f;
    for (int c = lane; c < ED; c += 32) {
      float v = (x[c] - mu) * rsg * glg[c] + glb[c];
      dot += gelu_erf(v) * gW2[c];
    }
    for (int off = 16; off; off >>= 1) dot += __shfl_xor(dot, off);
    if (lane == 0) {
      float gs = 1.f / (1.f + expf(-(dot + gb2[0])));
      float wsum = 0.f;
      for (int h = 0; h < NH; ++h) wsum += sPA[r * NH + h] * wsm[h];
      float conf = 1.f - (0.7f * wsum + 0.3f * gs);
      out_conf[row0 + r] = fminf(fmaxf(conf, 0.f), 1.f);
    }
  }
}

extern "C" void kernel_launch(void* const* d_in, const int* in_sizes, int n_in,
                              void* d_out, int out_size, void* d_ws, size_t ws_size,
                              hipStream_t stream) {
  (void)n_in; (void)out_size; (void)ws_size;
  const float* Xt  = (const float*)d_in[0];
  const float* Xi  = (const float*)d_in[1];
  const float* W1  = (const float*)d_in[2];
  const float* b1  = (const float*)d_in[3];
  const float* lg1 = (const float*)d_in[4];
  const float* lb1 = (const float*)d_in[5];
  const float* W2  = (const float*)d_in[6];
  const float* b2  = (const float*)d_in[7];
  const float* aw  = (const float*)d_in[8];
  const float* gW1 = (const float*)d_in[9];
  const float* gb1 = (const float*)d_in[10];
  const float* glg = (const float*)d_in[11];
  const float* glb = (const float*)d_in[12];
  const float* gW2 = (const float*)d_in[13];
  const float* gb2 = (const float*)d_in[14];

  const int batch = in_sizes[0] / ED;
  float* out_conf = (float*)d_out;
  float* out_pa   = out_conf + batch;
  bf16_t* wsb     = (bf16_t*)d_ws;            // needs WS_HALFS*2 = ~2.2 MB

  // 1) weight repack (bf16, fragment-linear) — tiny, L2-resident
  {
    long total = WS_HALFS;
    int blocks = (int)((total + 255) / 256);
    prep_weights<<<blocks, 256, 0, stream>>>(W1, W2, gW1, wsb);
  }

  // 2) fused forward
  const size_t smem =
      (size_t)ROWS * ED * 2 * sizeof(bf16_t)   // sXt + sXi (later sG)  64 KB
    + (size_t)ROWS * NC1 * sizeof(bf16_t)      // sHn                   64 KB
    + (size_t)ROWS * NH * sizeof(float);       // sPA        (~129 KB -> 2 WGs/WGP)

  hipFuncSetAttribute(reinterpret_cast<const void*>(fused_conflict_kernel),
                      hipFuncAttributeMaxDynamicSharedMemorySize, (int)smem);

  dim3 grid(batch / ROWS), block(NTHREADS);
  fused_conflict_kernel<<<grid, block, smem, stream>>>(
      Xt, Xi, b1, lg1, lb1, b2, aw, gb1, glg, glb, gW2, gb2,
      wsb, out_conf, out_pa);
}
